// MCA_12214886990440
// MI455X (gfx1250) — matrix operand order repaired and checked by reading, not compiled, not run on hardware
//
#include <hip/hip_runtime.h>
#include <math.h>

typedef _Float16 half_t;
typedef __attribute__((ext_vector_type(16))) _Float16 v16h;
typedef __attribute__((ext_vector_type(8)))  _Float16 v8h;
typedef __attribute__((ext_vector_type(8)))  float    v8f;
typedef __attribute__((ext_vector_type(4)))  unsigned int u32x4;
typedef __attribute__((ext_vector_type(8)))  int i32x8;
typedef __attribute__((ext_vector_type(4)))  int i32x4;

#define BB 8
#define CC 64
#define HH 128
#define WW 128
#define PP 131072L            // B*H*W
#define CP (64L * 131072L)    // C*P
#define LDS_STRIDE 72         // padded LDS row stride (halfs): 64 data + 8 pad

// ---------------------------------------------------------------------------
// WMMA helper (CDNA5 v_wmma_f32_16x16x32_f16)
// ---------------------------------------------------------------------------
__device__ __forceinline__ v8f wmma16(v16h a, v16h b, v8f c) {
  return __builtin_amdgcn_wmma_f32_16x16x32_f16(
      false, a, false, b, (short)0, c, false, false);
}

// A fragment: 16x32 f16, row-major with leading dim ldm (halfs).
// Lane l: m = l&15, K groups [ka..ka+7], [ka+16..ka+23], ka=(l<16)?0:8.
__device__ __forceinline__ v16h load_a_frag(const half_t* __restrict__ base,
                                            int ldm, int lane) {
  int m  = lane & 15;
  int ka = (lane < 16) ? 0 : 8;
  const half_t* p = base + (long)m * ldm + ka;
  v8h lo = *(const v8h*)(p);
  v8h hi = *(const v8h*)(p + 16);
  return __builtin_shufflevector(lo, hi, 0,1,2,3,4,5,6,7,8,9,10,11,12,13,14,15);
}

// B fragment: 32x16 f16 where each column n has K contiguous (B^T row-major),
// leading dim ldn (halfs). Lane l: n=l&15, K groups [kb..kb+7], [kb+16..kb+23], kb=(l<16)?0:8.
__device__ __forceinline__ v16h load_b_frag(const half_t* __restrict__ base,
                                            int ldn, int lane) {
  int n  = lane & 15;
  int kb = (lane < 16) ? 0 : 8;
  const half_t* p = base + (long)n * ldn + kb;
  v8h lo = *(const v8h*)(p);
  v8h hi = *(const v8h*)(p + 16);
  return __builtin_shufflevector(lo, hi, 0,1,2,3,4,5,6,7,8,9,10,11,12,13,14,15);
}

// ---------------------------------------------------------------------------
// TDM: DMA `nrows` rows of 64 halfs (pixel-major activations) from global
// into LDS with a 16B pad after every 128B row  -> LDS row stride = 72 halfs.
// Descriptor built per CDNA5 ISA.
// ---------------------------------------------------------------------------
#if __has_builtin(__builtin_amdgcn_tensor_load_to_lds)
#define USE_TDM 1
#else
#define USE_TDM 0
#endif

__device__ __forceinline__ void tdm_load_rows(const half_t* gsrc,
                                              unsigned lds_byte_addr,
                                              int nrows) {
#if USE_TDM
  unsigned long long ga = (unsigned long long)gsrc;
  u32x4 g0;
  g0[0] = 1u;                                         // count=1 (valid D#)
  g0[1] = lds_byte_addr;                              // lds_addr [63:32]
  g0[2] = (unsigned)(ga & 0xFFFFFFFFu);               // global_addr lo
  g0[3] = (unsigned)((ga >> 32) & 0x1FFFFFFu)         // global_addr [56:32]
        | (2u << 30);                                 // type=2 ("image")
  i32x8 g1;
  g1[0] = (1 << 16)      // data_size = 2 bytes
        | (1 << 20)      // pad_enable
        | (4 << 22)      // pad_interval: 32 dwords = 128B
        | (3 << 25);     // pad_amount:   4 dwords = 16B
  unsigned td0 = 64u;                 // tensor dim0 (elems/row)
  unsigned td1 = (unsigned)PP;        // tensor dim1 (total rows, OOB bound)
  g1[1] = (int)((td0 & 0xFFFFu) << 16);
  g1[2] = (int)(((td0 >> 16) & 0xFFFFu) | ((td1 & 0xFFFFu) << 16));
  g1[3] = (int)(((td1 >> 16) & 0xFFFFu) | (64u << 16));   // tile_dim0 = 64
  g1[4] = (int)(nrows & 0xFFFF);                          // tile_dim1 = nrows
  g1[5] = 64;                                             // dim0 stride = 64
  g1[6] = 0;
  g1[7] = 0;
  i32x4 z4 = {0, 0, 0, 0};
#if defined(__clang_major__) && __clang_major__ >= 23
  i32x8 z8 = {0, 0, 0, 0, 0, 0, 0, 0};
  __builtin_amdgcn_tensor_load_to_lds(g0, g1, z4, z4, z8, 0);
#else
  __builtin_amdgcn_tensor_load_to_lds(g0, g1, z4, z4, 0);
#endif
#else
  (void)gsrc; (void)lds_byte_addr; (void)nrows;
#endif
}

// Manual fallback copy (row-straight, packed b128 both sides).
__device__ __forceinline__ void stage_rows_manual(const half_t* __restrict__ gsrc,
                                                  half_t* Bt, int lo, int nrows,
                                                  int tid) {
  for (int r = tid; r < nrows; r += 256) {
    const half_t* s = gsrc + (long)r * 64;
    half_t* d = Bt + (lo + r) * LDS_STRIDE;
#pragma unroll
    for (int q = 0; q < 8; ++q) *(v8h*)(d + q * 8) = *(const v8h*)(s + q * 8);
  }
}

// ---------------------------------------------------------------------------
// 1) BN fold: x (NCHW f32) -> x_h, x_w pixel-major [P][C] f16
// ---------------------------------------------------------------------------
__global__ __launch_bounds__(256) void k_prep(
    const float* __restrict__ x,
    const float* __restrict__ g1, const float* __restrict__ b1,
    const float* __restrict__ m1, const float* __restrict__ v1,
    const float* __restrict__ g2, const float* __restrict__ b2,
    const float* __restrict__ m2, const float* __restrict__ v2,
    half_t* __restrict__ xh, half_t* __restrict__ xw) {
  long i = (long)blockIdx.x * 256 + threadIdx.x;
  if (i >= (long)BB * CC * HH * WW) return;
  int w = (int)(i & 127);
  int h = (int)((i >> 7) & 127);
  int c = (int)((i >> 14) & 63);
  int b = (int)(i >> 20);
  float xv = x[i];
  float a1 = g1[c] * rsqrtf(v1[c] + 1e-5f);
  float a2 = g2[c] * rsqrtf(v2[c] + 1e-5f);
  long p = ((long)b * HH + h) * WW + w;
  xh[p * 64 + c] = (half_t)((xv - m1[c]) * a1 + b1[c]);
  xw[p * 64 + c] = (half_t)((xv - m2[c]) * a2 + b2[c]);
}

// ---------------------------------------------------------------------------
// 2) Weight prep: merge 7/11/21 taps, f16-convert 1x1 weights, sum biases.
// ---------------------------------------------------------------------------
__global__ __launch_bounds__(256) void k_prepw(
    const float* __restrict__ w7h,  const float* __restrict__ w11h, const float* __restrict__ w21h,
    const float* __restrict__ w7w,  const float* __restrict__ w11w, const float* __restrict__ w21w,
    const float* __restrict__ b7h,  const float* __restrict__ b11h, const float* __restrict__ b21h,
    const float* __restrict__ b7w,  const float* __restrict__ b11w, const float* __restrict__ b21w,
    const float* __restrict__ hqkvw, const float* __restrict__ wqkvw,
    const float* __restrict__ houtw, const float* __restrict__ woutw,
    half_t* __restrict__ weffh, half_t* __restrict__ weffw,
    float* __restrict__ biash,  float* __restrict__ biasw,
    half_t* __restrict__ hqkvh, half_t* __restrict__ wqkvh,
    half_t* __restrict__ houth, half_t* __restrict__ wouth) {
  int i = blockIdx.x * 256 + threadIdx.x;
  if (i < 21 * 4096) {
    int t  = i / 4096;
    int oi = i - t * 4096;      // o*64 + ic
    int dy = t - 10;
    float sh = w21h[(long)oi * 21 + t];
    float sw = w21w[(long)oi * 21 + t];
    if (dy >= -5 && dy <= 5) { sh += w11h[(long)oi * 11 + (dy + 5)];
                               sw += w11w[(long)oi * 11 + (dy + 5)]; }
    if (dy >= -3 && dy <= 3) { sh += w7h[(long)oi * 7 + (dy + 3)];
                               sw += w7w[(long)oi * 7 + (dy + 3)]; }
    weffh[(long)t * 4096 + oi] = (half_t)sh;
    weffw[(long)t * 4096 + oi] = (half_t)sw;
  }
  if (i < 12288) { hqkvh[i] = (half_t)hqkvw[i]; wqkvh[i] = (half_t)wqkvw[i]; }
  if (i < 4096)  { houth[i] = (half_t)houtw[i]; wouth[i] = (half_t)woutw[i]; }
  if (i < 64)    { biash[i] = b7h[i] + b11h[i] + b21h[i];
                   biasw[i] = b7w[i] + b11w[i] + b21w[i]; }
}

// ---------------------------------------------------------------------------
// 3) 21-tap strip conv: per image row, sum of 64x64 GEMMs over shifted rows.
//    TDM stages each shifted row tile (pixel-major -> straight copy) into
//    LDS with padded stride; W-dir edges zeroed with ds_store_b128.
// ---------------------------------------------------------------------------
__global__ __launch_bounds__(256) void k_stripconv(
    const half_t* __restrict__ src, const half_t* __restrict__ weff,
    const float* __restrict__ bias, half_t* __restrict__ dst, int dir) {
  __shared__ half_t Bt[128 * LDS_STRIDE];
  int bh = blockIdx.x;               // b*H + h
  long rowbase = (long)bh * WW;
  int h = bh & (HH - 1);
  int tid = threadIdx.x;
  int lane = tid & 31;
  int wv = tid >> 5;
  unsigned lds0 = (unsigned)(unsigned long long)&Bt[0];

  v8f acc[4] = {};
  for (int t = 0; t < 21; ++t) {
    int off = t - 10;
    bool rowvalid = true;
    long gstart;
    int lo, nrows;
    if (dir == 0) {
      int hh = h + off;
      rowvalid = (hh >= 0 && hh < HH);
      gstart = rowbase + (long)off * WW;
      lo = 0; nrows = 128;
    } else {
      lo = (off < 0) ? -off : 0;
      gstart = rowbase + ((off > 0) ? off : 0);
      nrows = 128 - ((off < 0) ? -off : off);
    }
    __syncthreads();                      // readers of previous tile done
    if (rowvalid) {
#if USE_TDM
      if (wv == 0) {
        tdm_load_rows(src + gstart * 64, lds0 + (unsigned)lo * (LDS_STRIDE * 2), nrows);
        __builtin_amdgcn_s_wait_tensorcnt(0);
      }
#else
      stage_rows_manual(src + gstart * 64, Bt, lo, nrows, tid);
#endif
      if (dir == 1 && tid < 128) {        // zero invalid edge rows
        int col = tid + off;
        if (col < 0 || col >= WW) {
          v8h z = {};
          half_t* row = Bt + tid * LDS_STRIDE;
#pragma unroll
          for (int q = 0; q < 8; ++q) *(v8h*)(row + q * 8) = z;
        }
      }
    }
    __syncthreads();
    if (!rowvalid) continue;              // uniform across block
    const half_t* wbase = weff + (long)t * 4096;
#pragma unroll
    for (int kh = 0; kh < 2; ++kh) {
      v16h bfrag = load_b_frag(Bt + (wv * 16) * LDS_STRIDE + kh * 32, LDS_STRIDE, lane);
#pragma unroll
      for (int mt = 0; mt < 4; ++mt) {
        v16h afrag = load_a_frag(wbase + (mt * 16) * 64 + kh * 32, 64, lane);
        acc[mt] = wmma16(afrag, bfrag, acc[mt]);
      }
    }
  }
  // packed b128 stores, dst pixel-major [P][64]
  int n = lane & 15;
  int moff = (lane < 16) ? 0 : 8;
  long prow = (rowbase + wv * 16 + n) * 64;
#pragma unroll
  for (int mt = 0; mt < 4; ++mt) {
    v8h hv;
#pragma unroll
    for (int r = 0; r < 8; ++r) {
      int o = mt * 16 + moff + r;
      hv[r] = (half_t)(acc[mt][r] + bias[o]);
    }
    *(v8h*)(dst + prow + mt * 16 + moff) = hv;
  }
}

// ---------------------------------------------------------------------------
// 4) qkv projection: (192x64) @ (64 x P), B frags direct from pixel-major src.
//    q,k -> channel-major [nh*8+d][p]; v -> to_wdh Vt[(b*8+nh)][w][d*128+h].
// ---------------------------------------------------------------------------
__global__ __launch_bounds__(256) void k_qkv(
    const half_t* __restrict__ src, const half_t* __restrict__ wmat,
    const float* __restrict__ bias,
    half_t* __restrict__ qdst, half_t* __restrict__ kdst, half_t* __restrict__ vtdst) {
  long pbase = (long)blockIdx.x * 128;
  int ocbase = blockIdx.y * 64;
  int tid = threadIdx.x, lane = tid & 31, wv = tid >> 5;

  v8f acc[4] = {};
#pragma unroll
  for (int kh = 0; kh < 2; ++kh) {
    v16h bfrag = load_b_frag(src + (pbase + wv * 16) * 64 + kh * 32, 64, lane);
#pragma unroll
    for (int mt = 0; mt < 4; ++mt) {
      v16h afrag = load_a_frag(wmat + (long)(ocbase + mt * 16) * 64 + kh * 32, 64, lane);
      acc[mt] = wmma16(afrag, bfrag, acc[mt]);
    }
  }
  int n = lane & 15;
  int moff = (lane < 16) ? 0 : 8;
  long pcol = pbase + wv * 16 + n;
  int b  = (int)(pcol >> 14);
  int hh = (int)((pcol >> 7) & 127);
  int w  = (int)(pcol & 127);
#pragma unroll
  for (int mt = 0; mt < 4; ++mt)
#pragma unroll
    for (int r = 0; r < 8; ++r) {
      int oc = ocbase + mt * 16 + moff + r;
      half_t hv = (half_t)(acc[mt][r] + bias[oc]);
      int nh = oc / 24, j = oc - nh * 24;
      int sel = j >> 3, d = j & 7;
      if (sel == 0)      qdst[(long)(nh * 8 + d) * PP + pcol] = hv;
      else if (sel == 1) kdst[(long)(nh * 8 + d) * PP + pcol] = hv;
      else vtdst[(long)(b * 8 + nh) * 131072 + (long)w * 1024 + d * 128 + hh] = hv;
    }
}

// ---------------------------------------------------------------------------
// 5) Attention GEMM1: attn[bnh][m=dh_k][n=dh_q], contraction over W.
// ---------------------------------------------------------------------------
__global__ __launch_bounds__(256) void k_attn1(
    const half_t* __restrict__ Ksrc, const half_t* __restrict__ Qsrc,
    half_t* __restrict__ attn) {
  int nblk = blockIdx.x, mblk = blockIdx.y, bnh = blockIdx.z;
  int b = bnh >> 3, nh = bnh & 7;
  int tid = threadIdx.x, lane = tid & 31, wv = tid >> 5;

  int ntile = nblk * 128 + wv * 16;
  int dn = ntile >> 7, hn0 = ntile & 127;
  int m0 = mblk * 64;
  const float scale = 8.0f * 0.08838834764831845f;  // D * H^-0.5

  v8f acc[4] = {};
#pragma unroll
  for (int kc = 0; kc < 4; ++kc) {
    const half_t* bbase = Qsrc + (long)(nh * 8 + dn) * PP
                        + ((long)b * 128 + hn0) * 128 + kc * 32;
    v16h bfrag = load_b_frag(bbase, 128, lane);
#pragma unroll
    for (int mt = 0; mt < 4; ++mt) {
      int m = m0 + mt * 16;
      const half_t* abase = Ksrc + (long)(nh * 8 + (m >> 7)) * PP
                          + ((long)b * 128 + (m & 127)) * 128 + kc * 32;
      v16h afrag = load_a_frag(abase, 128, lane);
      acc[mt] = wmma16(afrag, bfrag, acc[mt]);
    }
  }
  int n = ntile + (lane & 15);
  int moff = (lane < 16) ? 0 : 8;
  half_t* arow = attn + (long)bnh * 1024 * 1024;
#pragma unroll
  for (int mt = 0; mt < 4; ++mt)
#pragma unroll
    for (int r = 0; r < 8; ++r) {
      int m = m0 + mt * 16 + moff + r;
      arow[(long)m * 1024 + n] = (half_t)(acc[mt][r] * scale);
    }
}

// ---------------------------------------------------------------------------
// 6) Attention GEMM2: out[w][n=dh] = sum_m Vt[w][m]*attn[n][m];
//    result -> pixel-major ocm[p][c].
// ---------------------------------------------------------------------------
__global__ __launch_bounds__(256) void k_attn2(
    const half_t* __restrict__ Vt, const half_t* __restrict__ attn,
    half_t* __restrict__ ocm) {
  int nblk = blockIdx.x, mblk = blockIdx.y, bnh = blockIdx.z;
  int b = bnh >> 3, nh = bnh & 7;
  int tid = threadIdx.x, lane = tid & 31, wv = tid >> 5;

  int ntile = nblk * 128 + wv * 16;
  int m0 = mblk * 64;
  const half_t* vbase0 = Vt + (long)bnh * 131072;
  const half_t* abase0 = attn + (long)bnh * 1024 * 1024;

  v8f acc[4] = {};
  for (int kc = 0; kc < 32; ++kc) {
    v16h bfrag = load_b_frag(abase0 + (long)ntile * 1024 + kc * 32, 1024, lane);
#pragma unroll
    for (int mt = 0; mt < 4; ++mt) {
      v16h afrag = load_a_frag(vbase0 + (long)(m0 + mt * 16) * 1024 + kc * 32, 1024, lane);
      acc[mt] = wmma16(afrag, bfrag, acc[mt]);
    }
  }
  int n = ntile + (lane & 15);
  int moff = (lane < 16) ? 0 : 8;
  int c = nh * 8 + (n >> 7);
  long prow = ((long)b * 128 + (n & 127)) * 128;
#pragma unroll
  for (int mt = 0; mt < 4; ++mt)
#pragma unroll
    for (int r = 0; r < 8; ++r) {
      int w = m0 + mt * 16 + moff + r;
      ocm[(prow + w) * 64 + c] = (half_t)acc[mt][r];
    }
}

// ---------------------------------------------------------------------------
// 7) Output projection (64x64), f32 accumulate into gate buffer G[p][c].
// ---------------------------------------------------------------------------
__global__ __launch_bounds__(256) void k_proj(
    const half_t* __restrict__ src, const half_t* __restrict__ wmat,
    const float* __restrict__ bias, float* __restrict__ G, int accum) {
  long pbase = (long)blockIdx.x * 128;
  int tid = threadIdx.x, lane = tid & 31, wv = tid >> 5;

  v8f acc[4] = {};
#pragma unroll
  for (int kh = 0; kh < 2; ++kh) {
    v16h bfrag = load_b_frag(src + (pbase + wv * 16) * 64 + kh * 32, 64, lane);
#pragma unroll
    for (int mt = 0; mt < 4; ++mt) {
      v16h afrag = load_a_frag(wmat + (long)(mt * 16) * 64 + kh * 32, 64, lane);
      acc[mt] = wmma16(afrag, bfrag, acc[mt]);
    }
  }
  int n = lane & 15;
  int moff = (lane < 16) ? 0 : 8;
  long pcol = pbase + wv * 16 + n;
#pragma unroll
  for (int mt = 0; mt < 4; ++mt)
#pragma unroll
    for (int r = 0; r < 8; ++r) {
      int o = mt * 16 + moff + r;
      float val = acc[mt][r] + bias[o];
      long idx = pcol * 64 + o;
      G[idx] = accum ? (G[idx] + val) : val;
    }
}

// ---------------------------------------------------------------------------
// 8) Gate: out = x * sigmoid(G)
// ---------------------------------------------------------------------------
__global__ __launch_bounds__(256) void k_gate(
    const float* __restrict__ x, const float* __restrict__ G,
    float* __restrict__ out) {
  long i = (long)blockIdx.x * 256 + threadIdx.x;
  if (i >= (long)BB * CC * HH * WW) return;
  int w = (int)(i & 127);
  int h = (int)((i >> 7) & 127);
  int c = (int)((i >> 14) & 63);
  int b = (int)(i >> 20);
  long p = ((long)b << 14) + ((long)h << 7) + w;
  float g = G[p * 64 + c];
  out[i] = x[i] * (1.0f / (1.0f + expf(-g)));
}

// ---------------------------------------------------------------------------
extern "C" void kernel_launch(void* const* d_in, const int* in_sizes, int n_in,
                              void* d_out, int out_size, void* d_ws, size_t ws_size,
                              hipStream_t stream) {
  (void)in_sizes; (void)n_in; (void)out_size; (void)ws_size;

  const float* x     = (const float*)d_in[0];
  const float* bn1_g = (const float*)d_in[1];
  const float* bn1_b = (const float*)d_in[2];
  const float* bn1_m = (const float*)d_in[3];
  const float* bn1_v = (const float*)d_in[4];
  const float* bn2_g = (const float*)d_in[5];
  const float* bn2_b = (const float*)d_in[6];
  const float* bn2_m = (const float*)d_in[7];
  const float* bn2_v = (const float*)d_in[8];
  const float* sc1_w0 = (const float*)d_in[9];
  const float* sc1_b0 = (const float*)d_in[10];
  const float* sc2_w0 = (const float*)d_in[11];
  const float* sc2_b0 = (const float*)d_in[12];
  const float* sc1_w1 = (const float*)d_in[13];
  const float* sc1_b1 = (const float*)d_in[14];
  const float* sc2_w1 = (const float*)d_in[15];
  const float* sc2_b1 = (const float*)d_in[16];
  const float* sc1_w2 = (const float*)d_in[17];
  const float* sc1_b2 = (const float*)d_in[18];
  const float* sc2_w2 = (const float*)d_in[19];
  const float* sc2_b2 = (const float*)d_in[20];
  const float* hqkv_w = (const float*)d_in[21];
  const float* hqkv_b = (const float*)d_in[22];
  const float* wqkv_w = (const float*)d_in[23];
  const float* wqkv_b = (const float*)d_in[24];
  const float* hout_w = (const float*)d_in[25];
  const float* hout_b = (const float*)d_in[26];
  const float* wout_w = (const float*)d_in[27];
  const float* wout_b = (const float*)d_in[28];

  char* base = (char*)d_ws;
  size_t off = 0;
  auto carve = [&](size_t bytes) -> void* {
    void* p = base + off;
    off += (bytes + 255) & ~(size_t)255;
    return p;
  };
  half_t* xh   = (half_t*)carve(CP * 2);
  half_t* xw   = (half_t*)carve(CP * 2);
  half_t* sch  = (half_t*)carve(CP * 2);
  half_t* scw  = (half_t*)carve(CP * 2);
  half_t* hq   = (half_t*)carve(CP * 2);
  half_t* hk   = (half_t*)carve(CP * 2);
  half_t* hvt  = (half_t*)carve(CP * 2);
  half_t* wq   = (half_t*)carve(CP * 2);
  half_t* wk   = (half_t*)carve(CP * 2);
  half_t* wvt  = (half_t*)carve(CP * 2);
  half_t* ocm  = (half_t*)carve(CP * 2);
  half_t* attn = (half_t*)carve((size_t)64 * 1024 * 1024 * 2);
  float*  G    = (float*)carve(CP * 4);
  half_t* weffh = (half_t*)carve(21 * 4096 * 2);
  half_t* weffw = (half_t*)carve(21 * 4096 * 2);
  half_t* hqkvh = (half_t*)carve(12288 * 2);
  half_t* wqkvh = (half_t*)carve(12288 * 2);
  half_t* houth = (half_t*)carve(4096 * 2);
  half_t* wouth = (half_t*)carve(4096 * 2);
  float*  biash = (float*)carve(64 * 4);
  float*  biasw = (float*)carve(64 * 4);

  const long NTOT = (long)BB * CC * HH * WW;  // 8388608

  k_prep<<<(NTOT + 255) / 256, 256, 0, stream>>>(
      x, bn1_g, bn1_b, bn1_m, bn1_v, bn2_g, bn2_b, bn2_m, bn2_v, xh, xw);

  k_prepw<<<(21 * 4096 + 255) / 256, 256, 0, stream>>>(
      sc1_w0, sc1_w1, sc1_w2, sc2_w0, sc2_w1, sc2_w2,
      sc1_b0, sc1_b1, sc1_b2, sc2_b0, sc2_b1, sc2_b2,
      hqkv_w, wqkv_w, hout_w, wout_w,
      weffh, weffw, biash, biasw, hqkvh, wqkvh, houth, wouth);

  k_stripconv<<<BB * HH, 256, 0, stream>>>(xh, weffh, biash, sch, 0);
  k_stripconv<<<BB * HH, 256, 0, stream>>>(xw, weffw, biasw, scw, 1);

  k_qkv<<<dim3(1024, 3), 256, 0, stream>>>(sch, hqkvh, hqkv_b, hq, hk, hvt);
  k_qkv<<<dim3(1024, 3), 256, 0, stream>>>(scw, wqkvh, wqkv_b, wq, wk, wvt);

  // W branch
  k_attn1<<<dim3(8, 16, 64), 256, 0, stream>>>(wk, hq, attn);
  k_attn2<<<dim3(8, 2, 64), 256, 0, stream>>>(wvt, attn, ocm);
  k_proj<<<1024, 256, 0, stream>>>(ocm, wouth, wout_b, G, 0);

  // H branch
  k_attn1<<<dim3(8, 16, 64), 256, 0, stream>>>(hk, wq, attn);
  k_attn2<<<dim3(8, 2, 64), 256, 0, stream>>>(hvt, attn, ocm);
  k_proj<<<1024, 256, 0, stream>>>(ocm, houth, hout_b, G, 1);

  k_gate<<<(NTOT + 255) / 256, 256, 0, stream>>>(x, G, (float*)d_out);
}
